// DiTBlock_CrossAttn_60206851555415
// MI455X (gfx1250) — matrix operand order repaired and checked by reading, not compiled, not run on hardware
//
#include <hip/hip_runtime.h>
#include <hip/hip_bf16.h>
#include <math.h>

typedef unsigned short u16;
typedef unsigned int   u32;

typedef __bf16 bf16t;
typedef bf16t v16bf __attribute__((ext_vector_type(16)));
typedef float v8f   __attribute__((ext_vector_type(8)));

__device__ __forceinline__ u16 f2bf(float f) {
    union { float f; u32 u; } v; v.f = f;
    u32 r = v.u + 0x7FFFu + ((v.u >> 16) & 1u);   // round-to-nearest-even
    return (u16)(r >> 16);
}

union Frag16 { v16bf v; uint4 q[2]; };

__device__ __forceinline__ v8f wmma_bf16(const Frag16& a, const Frag16& b, v8f c) {
    return __builtin_amdgcn_wmma_f32_16x16x32_bf16(false, a.v, false, b.v,
                                                   (short)0, c, false, false);
}

// Butterfly reductions across the 16-lane half-groups using ds_swizzle
// (group-of-32 mode, and_mask=0x1f, xor_mask in bits[14:10]).
__device__ __forceinline__ float red16_max(float v) {
    v = fmaxf(v, __int_as_float(__builtin_amdgcn_ds_swizzle(__float_as_int(v), 0x041f)));
    v = fmaxf(v, __int_as_float(__builtin_amdgcn_ds_swizzle(__float_as_int(v), 0x081f)));
    v = fmaxf(v, __int_as_float(__builtin_amdgcn_ds_swizzle(__float_as_int(v), 0x101f)));
    v = fmaxf(v, __int_as_float(__builtin_amdgcn_ds_swizzle(__float_as_int(v), 0x201f)));
    return v;
}
__device__ __forceinline__ float red16_sum(float v) {
    v += __int_as_float(__builtin_amdgcn_ds_swizzle(__float_as_int(v), 0x041f));
    v += __int_as_float(__builtin_amdgcn_ds_swizzle(__float_as_int(v), 0x081f));
    v += __int_as_float(__builtin_amdgcn_ds_swizzle(__float_as_int(v), 0x101f));
    v += __int_as_float(__builtin_amdgcn_ds_swizzle(__float_as_int(v), 0x201f));
    return v;
}

// ---------------------------------------------------------------------------
// Small helper kernels (fp32 paths)
// ---------------------------------------------------------------------------

// wt[n*K+k] = bf16(w[k*N+n])  : weight transpose+convert -> K-major bf16
__global__ void wtrans_kernel(const float* __restrict__ w, u16* __restrict__ wt,
                              int K, int N) {
    size_t idx = (size_t)blockIdx.x * 256 + threadIdx.x;
    if (idx >= (size_t)K * N) return;
    int k = (int)(idx % K);
    int n = (int)(idx / K);
    wt[idx] = f2bf(w[(size_t)k * N + n]);
}

__global__ void cvt_bf16_kernel(const float* __restrict__ in, u16* __restrict__ out,
                                size_t n) {
    size_t idx = (size_t)blockIdx.x * 256 + threadIdx.x;
    if (idx < n) out[idx] = f2bf(in[idx]);
}

// ada[b, 0:6144] = silu(c[b]) @ w_ada + b_ada   (B=8, tiny GEMM, fp32)
__global__ void ada_kernel(const float* __restrict__ c, const float* __restrict__ w,
                           const float* __restrict__ bias, float* __restrict__ ada) {
    int idx = blockIdx.x * 256 + threadIdx.x;       // 8*6144 total
    if (idx >= 8 * 6144) return;
    int b = idx / 6144, col = idx % 6144;
    const float* cr = c + (size_t)b * 1024;
    float acc = bias[col];
    for (int k = 0; k < 1024; ++k) {
        float cv = cr[k];
        float s  = cv / (1.0f + __expf(-cv));       // SiLU
        acc += s * w[(size_t)k * 6144 + col];
    }
    ada[idx] = acc;
}

// h = modulate(LN(x), shift, scale) -> bf16.  One 256-thread block per row (D=1024).
__global__ __launch_bounds__(256)
void ln_mod_kernel(const float* __restrict__ x, const float* __restrict__ ada,
                   u16* __restrict__ out, int rowsPerBatch,
                   int shiftOff, int scaleOff) {
    __shared__ float red[16];
    const int row = blockIdx.x;
    const int b   = row / rowsPerBatch;
    const float* xr = x + (size_t)row * 1024;
    float xl[4];
    *(float4*)xl = ((const float4*)xr)[threadIdx.x];
    float s = 0.f, ss = 0.f;
#pragma unroll
    for (int i = 0; i < 4; ++i) { s += xl[i]; ss += xl[i] * xl[i]; }
#pragma unroll
    for (int m = 1; m < 32; m <<= 1) {
        s  += __shfl_xor(s,  m, 32);
        ss += __shfl_xor(ss, m, 32);
    }
    const int wave = threadIdx.x >> 5, lane = threadIdx.x & 31;
    if (lane == 0) { red[wave] = s; red[8 + wave] = ss; }
    __syncthreads();
    float S = 0.f, SS = 0.f;
#pragma unroll
    for (int w = 0; w < 8; ++w) { S += red[w]; SS += red[8 + w]; }
    const float mu  = S * (1.0f / 1024.0f);
    const float var = SS * (1.0f / 1024.0f) - mu * mu;
    const float rsd = rsqrtf(var + 1e-6f);
    const float* adab = ada + (size_t)b * 6144;
    const int d0 = threadIdx.x * 4;
    ushort4 o;
    u16 ov[4];
#pragma unroll
    for (int i = 0; i < 4; ++i) {
        int d = d0 + i;
        float v = (xl[i] - mu) * rsd;
        v = v * (1.0f + adab[scaleOff + d]) + adab[shiftOff + d];
        ov[i] = f2bf(v);
    }
    o.x = ov[0]; o.y = ov[1]; o.z = ov[2]; o.w = ov[3];
    *(ushort4*)(out + (size_t)row * 1024 + d0) = o;
}

// Repack [B,Rows,W] bf16 (head-interleaved cols at colOff) to per-head layout:
//   vmode==0: out[b,h,r,d]  (Q/K)        vmode==1: out[b,h,d,r]  (V transposed)
__global__ void repack_kernel(const u16* __restrict__ in, u16* __restrict__ out,
                              int Bn, int Rows, int W, int colOff, int vmode) {
    size_t idx = (size_t)blockIdx.x * 256 + threadIdx.x;
    size_t total = (size_t)Bn * Rows * 1024;
    if (idx >= total) return;
    int d = (int)(idx & 63);
    int h = (int)((idx >> 6) & 15);
    size_t t = idx >> 10;
    int r = (int)(t % Rows);
    int b = (int)(t / Rows);
    u16 v = in[(size_t)(b * Rows + r) * W + colOff + h * 64 + d];
    if (!vmode) out[((size_t)(b * 16 + h) * Rows + r) * 64 + d] = v;
    else        out[((size_t)(b * 16 + h) * 64 + d) * Rows + r] = v;
}

// ---------------------------------------------------------------------------
// WMMA GEMM: C[M,N] = A[M,K](bf16) @ Wt[N,K](bf16)^T + bias, fused epilogue.
// Block = 256 threads (8 waves). Tile 128x256, K-step 32.
// Waves arranged 2x4; each wave owns a 64x64 sub-tile -> 4x4 WMMA fragments
// (16 WMMA per K-step from 8 fragment loads: 16 LDS bytes/lane per WMMA).
// ---------------------------------------------------------------------------
#define EPI_F32    0
#define EPI_BF16   1
#define EPI_RESADD 2
#define EPI_GELU   3

template<int EPI>
__global__ __launch_bounds__(256)
void gemm_bf16_kernel(const u16* __restrict__ A, const u16* __restrict__ Wt,
                      const float* __restrict__ bias, const float* __restrict__ res,
                      float* __restrict__ outF, u16* __restrict__ outB,
                      int M, int N, int K) {
    __shared__ __align__(16) u16 As[128 * 48];
    __shared__ __align__(16) u16 Bs[256 * 48];
    const int tid  = threadIdx.x;
    const int lane = tid & 31, wave = tid >> 5;
    const int m16  = lane & 15, hi = lane >> 4;
    const int wm   = wave >> 2, wn = wave & 3;
    const int rBase = blockIdx.y * 128;
    const int cBase = blockIdx.x * 256;

    v8f c[4][4];
#pragma unroll
    for (int i = 0; i < 4; ++i)
#pragma unroll
        for (int j = 0; j < 4; ++j)
#pragma unroll
            for (int r = 0; r < 8; ++r) c[i][j][r] = 0.0f;

    const int cpRow = tid >> 1, cpSeg = tid & 1;

    for (int k0 = 0; k0 < K; k0 += 32) {
        __syncthreads();
        {
            const uint4* sa = (const uint4*)(A + (size_t)(rBase + cpRow) * K + k0 + cpSeg * 16);
            uint4 a0 = sa[0], a1 = sa[1];
            uint4* da = (uint4*)&As[cpRow * 48 + cpSeg * 16];
            da[0] = a0; da[1] = a1;
#pragma unroll
            for (int it = 0; it < 2; ++it) {
                const int row = it * 128 + cpRow;
                const uint4* sb = (const uint4*)(Wt + (size_t)(cBase + row) * K + k0 + cpSeg * 16);
                uint4 b0 = sb[0], b1 = sb[1];
                uint4* db = (uint4*)&Bs[row * 48 + cpSeg * 16];
                db[0] = b0; db[1] = b1;
            }
        }
        __syncthreads();

        Frag16 a[4], b[4];
#pragma unroll
        for (int i = 0; i < 4; ++i) {
            int row = wm * 64 + i * 16 + m16;
            a[i].q[0] = *(const uint4*)&As[row * 48 + hi * 8];
            a[i].q[1] = *(const uint4*)&As[row * 48 + 16 + hi * 8];
        }
#pragma unroll
        for (int j = 0; j < 4; ++j) {
            int col = wn * 64 + j * 16 + m16;
            const uint4* p = (const uint4*)&Bs[col * 48 + hi * 8];
            b[j].q[0] = p[0];
            b[j].q[1] = p[2];
        }
#pragma unroll
        for (int i = 0; i < 4; ++i)
#pragma unroll
            for (int j = 0; j < 4; ++j)
                c[i][j] = wmma_bf16(a[i], b[j], c[i][j]);
    }

#pragma unroll
    for (int i = 0; i < 4; ++i)
#pragma unroll
        for (int j = 0; j < 4; ++j) {
            const int gn = cBase + wn * 64 + j * 16 + m16;
            const float bv = bias[gn];
#pragma unroll
            for (int r = 0; r < 8; ++r) {
                const int gm = rBase + wm * 64 + i * 16 + r + 8 * hi;
                const size_t o = (size_t)gm * N + gn;
                float v = c[i][j][r] + bv;
                if (EPI == EPI_F32)    outF[o] = v;
                if (EPI == EPI_BF16)   outB[o] = f2bf(v);
                if (EPI == EPI_RESADD) outF[o] = res[o] + v;
                if (EPI == EPI_GELU)   outB[o] = f2bf(0.5f * v * (1.0f + erff(v * 0.70710678118654752f)));
            }
        }
}

// ---------------------------------------------------------------------------
// Flash attention, bf16 WMMA, online softmax.
// qb,kb: [B,H,Nx,64]; vt: [B,H,64,Nkv]; out: [B,Nq,H*64] bf16.
// Block = 128 threads (4 waves); wave owns 16 query rows; key blocks of 32.
// ---------------------------------------------------------------------------
__global__ __launch_bounds__(128)
void flash_kernel(const u16* __restrict__ qb, const u16* __restrict__ kb,
                  const u16* __restrict__ vt, u16* __restrict__ out,
                  int Nq, int Nkv) {
    __shared__ __align__(16) u16 Pl[4][16 * 48];
    const int tid  = threadIdx.x;
    const int lane = tid & 31, wave = tid >> 5;
    const int m    = lane & 15, hi = lane >> 4;
    const int b = blockIdx.z, h = blockIdx.y;
    const int q0 = blockIdx.x * 64 + wave * 16;
    const u16* Q  = qb + (size_t)(b * 16 + h) * Nq  * 64;
    const u16* Kp = kb + (size_t)(b * 16 + h) * Nkv * 64;
    const u16* Vt = vt + (size_t)(b * 16 + h) * 64 * Nkv;
    u16* Pw = &Pl[wave][0];

    Frag16 qf[2];
#pragma unroll
    for (int t = 0; t < 2; ++t) {
        const u16* p = Q + (size_t)(q0 + m) * 64 + t * 32 + hi * 8;
        qf[t].q[0] = *(const uint4*)p;
        qf[t].q[1] = *(const uint4*)(p + 16);
    }

    float mi[8], li[8];
    v8f acc[4];
#pragma unroll
    for (int r = 0; r < 8; ++r) { mi[r] = -1e30f; li[r] = 0.f; }
#pragma unroll
    for (int f = 0; f < 4; ++f)
#pragma unroll
        for (int r = 0; r < 8; ++r) acc[f][r] = 0.f;

    for (int j0 = 0; j0 < Nkv; j0 += 32) {
        v8f s01[2];
#pragma unroll
        for (int t = 0; t < 2; ++t) {
            v8f s;
#pragma unroll
            for (int r = 0; r < 8; ++r) s[r] = 0.f;
#pragma unroll
            for (int ks = 0; ks < 2; ++ks) {
                Frag16 bf;
                const u16* p = Kp + (size_t)(j0 + t * 16 + m) * 64 + ks * 32 + hi * 8;
                bf.q[0] = *(const uint4*)p;
                bf.q[1] = *(const uint4*)(p + 16);
                s = wmma_bf16(qf[ks], bf, s);
            }
            s01[t] = s;
        }
        float p0[8], p1[8];
#pragma unroll
        for (int r = 0; r < 8; ++r) {
            float v0 = s01[0][r] * 0.125f;          // 1/sqrt(64)
            float v1 = s01[1][r] * 0.125f;
            float mx = red16_max(fmaxf(v0, v1));
            float nm = fmaxf(mi[r], mx);
            float cf = __expf(mi[r] - nm);
            float e0 = __expf(v0 - nm), e1 = __expf(v1 - nm);
            float rs = red16_sum(e0 + e1);
            li[r] = li[r] * cf + rs;
            mi[r] = nm;
            p0[r] = e0; p1[r] = e1;
#pragma unroll
            for (int f = 0; f < 4; ++f) acc[f][r] = acc[f][r] * cf;
        }
        // P (C-layout) -> LDS -> A-layout fragment
#pragma unroll
        for (int r = 0; r < 8; ++r) {
            int rr = r + 8 * hi;
            Pw[rr * 48 + m]      = f2bf(p0[r]);
            Pw[rr * 48 + 16 + m] = f2bf(p1[r]);
        }
        asm volatile("s_wait_dscnt 0x0" ::: "memory");
        Frag16 pf;
        {
            const u16* p = Pw + m * 48 + hi * 8;
            pf.q[0] = *(const uint4*)p;
            pf.q[1] = *(const uint4*)(p + 16);
        }
#pragma unroll
        for (int f = 0; f < 4; ++f) {
            Frag16 vf;
            const u16* p = Vt + (size_t)(f * 16 + m) * Nkv + j0 + hi * 8;
            vf.q[0] = *(const uint4*)p;
            vf.q[1] = *(const uint4*)(p + 16);
            acc[f] = wmma_bf16(pf, vf, acc[f]);
        }
    }

#pragma unroll
    for (int r = 0; r < 8; ++r) {
        const float inv = 1.0f / li[r];
        const int qrow = q0 + r + 8 * hi;
#pragma unroll
        for (int f = 0; f < 4; ++f) {
            out[(size_t)(b * Nq + qrow) * 1024 + h * 64 + f * 16 + m] = f2bf(acc[f][r] * inv);
        }
    }
}

// ---------------------------------------------------------------------------
// Launcher
// ---------------------------------------------------------------------------
extern "C" void kernel_launch(void* const* d_in, const int* in_sizes, int n_in,
                              void* d_out, int out_size, void* d_ws, size_t ws_size,
                              hipStream_t stream) {
    (void)in_sizes; (void)n_in; (void)out_size; (void)ws_size;
    const float* x     = (const float*)d_in[0];
    const float* cvec  = (const float*)d_in[1];
    const float* ctx   = (const float*)d_in[2];
    const float* w_qkv = (const float*)d_in[3];  const float* b_qkv = (const float*)d_in[4];
    const float* w_so  = (const float*)d_in[5];  const float* b_so  = (const float*)d_in[6];
    const float* w_cq  = (const float*)d_in[7];  const float* b_cq  = (const float*)d_in[8];
    const float* w_ck  = (const float*)d_in[9];  const float* b_ck  = (const float*)d_in[10];
    const float* w_cv  = (const float*)d_in[11]; const float* b_cv  = (const float*)d_in[12];
    const float* w_co  = (const float*)d_in[13]; const float* b_co  = (const float*)d_in[14];
    const float* w1    = (const float*)d_in[15]; const float* b1    = (const float*)d_in[16];
    const float* w2    = (const float*)d_in[17]; const float* b2    = (const float*)d_in[18];
    const float* w_ada = (const float*)d_in[19]; const float* b_ada = (const float*)d_in[20];
    float* outP = (float*)d_out;

    const size_t MB = 1024 * 1024;
    char* ws = (char*)d_ws;
    size_t off = 0;
    auto alloc = [&](size_t bytes) -> char* {
        char* p = ws + off;
        off += (bytes + 255) & ~(size_t)255;
        return p;
    };

    u16* wqkvT = (u16*)alloc((size_t)3072 * 1024 * 2);
    u16* wsoT  = (u16*)alloc((size_t)1024 * 1024 * 2);
    u16* wcqT  = (u16*)alloc((size_t)1024 * 1024 * 2);
    u16* wckT  = (u16*)alloc((size_t)1024 * 512 * 2);
    u16* wcvT  = (u16*)alloc((size_t)1024 * 512 * 2);
    u16* wcoT  = (u16*)alloc((size_t)1024 * 1024 * 2);
    u16* w1T   = (u16*)alloc((size_t)4096 * 1024 * 2);
    u16* w2T   = (u16*)alloc((size_t)1024 * 4096 * 2);
    float* ada = (float*)alloc((size_t)8 * 6144 * 4);
    u16* ctxb  = (u16*)alloc((size_t)8 * 256 * 512 * 2);
    float* xa  = (float*)alloc((size_t)8192 * 1024 * 4);
    u16* hb    = (u16*)alloc((size_t)8192 * 1024 * 2);
    u16* attn  = (u16*)alloc((size_t)8192 * 1024 * 2);
    char* pool = alloc(96 * MB);
    // self-attn layout in pool
    u16* qkvb = (u16*)(pool);
    u16* qb   = (u16*)(pool + 48 * MB);
    u16* kbuf = (u16*)(pool + 64 * MB);
    u16* vtb  = (u16*)(pool + 80 * MB);
    // cross-attn layout in pool (reused after self-attn)
    u16* cqraw = (u16*)(pool);
    u16* cqb   = (u16*)(pool + 16 * MB);
    u16* ckraw = (u16*)(pool + 32 * MB);
    u16* ckb   = (u16*)(pool + 40 * MB);
    u16* cvraw = (u16*)(pool + 48 * MB);
    u16* cvtb  = (u16*)(pool + 56 * MB);
    // FFN hidden (reused after cross-attn)
    u16* g = (u16*)(pool);

    auto wt = [&](const float* w, u16* dst, int K, int N) {
        size_t tot = (size_t)K * N;
        wtrans_kernel<<<(unsigned)((tot + 255) / 256), 256, 0, stream>>>(w, dst, K, N);
    };
    // 1) weight prep (transpose+bf16)
    wt(w_qkv, wqkvT, 1024, 3072);
    wt(w_so,  wsoT,  1024, 1024);
    wt(w_cq,  wcqT,  1024, 1024);
    wt(w_ck,  wckT,  512,  1024);
    wt(w_cv,  wcvT,  512,  1024);
    wt(w_co,  wcoT,  1024, 1024);
    wt(w1,    w1T,   1024, 4096);
    wt(w2,    w2T,   4096, 1024);
    cvt_bf16_kernel<<<(8 * 256 * 512) / 256, 256, 0, stream>>>(ctx, ctxb, (size_t)8 * 256 * 512);

    // 2) adaLN
    ada_kernel<<<(8 * 6144) / 256, 256, 0, stream>>>(cvec, w_ada, b_ada, ada);

    // 3) self-attention
    ln_mod_kernel<<<8192, 256, 0, stream>>>(x, ada, hb, 1024, 0, 1024);
    gemm_bf16_kernel<EPI_BF16><<<dim3(3072 / 256, 8192 / 128), 256, 0, stream>>>(
        hb, wqkvT, b_qkv, nullptr, nullptr, qkvb, 8192, 3072, 1024);
    repack_kernel<<<(8 * 1024 * 1024) / 256, 256, 0, stream>>>(qkvb, qb,   8, 1024, 3072, 0,    0);
    repack_kernel<<<(8 * 1024 * 1024) / 256, 256, 0, stream>>>(qkvb, kbuf, 8, 1024, 3072, 1024, 0);
    repack_kernel<<<(8 * 1024 * 1024) / 256, 256, 0, stream>>>(qkvb, vtb,  8, 1024, 3072, 2048, 1);
    flash_kernel<<<dim3(1024 / 64, 16, 8), 128, 0, stream>>>(qb, kbuf, vtb, attn, 1024, 1024);
    gemm_bf16_kernel<EPI_RESADD><<<dim3(1024 / 256, 8192 / 128), 256, 0, stream>>>(
        attn, wsoT, b_so, x, xa, nullptr, 8192, 1024, 1024);

    // 4) cross-attention
    ln_mod_kernel<<<8192, 256, 0, stream>>>(xa, ada, hb, 1024, 2048, 3072);
    gemm_bf16_kernel<EPI_BF16><<<dim3(1024 / 256, 8192 / 128), 256, 0, stream>>>(
        hb, wcqT, b_cq, nullptr, nullptr, cqraw, 8192, 1024, 1024);
    repack_kernel<<<(8 * 1024 * 1024) / 256, 256, 0, stream>>>(cqraw, cqb, 8, 1024, 1024, 0, 0);
    gemm_bf16_kernel<EPI_BF16><<<dim3(1024 / 256, 2048 / 128), 256, 0, stream>>>(
        ctxb, wckT, b_ck, nullptr, nullptr, ckraw, 2048, 1024, 512);
    repack_kernel<<<(8 * 256 * 1024) / 256, 256, 0, stream>>>(ckraw, ckb, 8, 256, 1024, 0, 0);
    gemm_bf16_kernel<EPI_BF16><<<dim3(1024 / 256, 2048 / 128), 256, 0, stream>>>(
        ctxb, wcvT, b_cv, nullptr, nullptr, cvraw, 2048, 1024, 512);
    repack_kernel<<<(8 * 256 * 1024) / 256, 256, 0, stream>>>(cvraw, cvtb, 8, 256, 1024, 0, 1);
    flash_kernel<<<dim3(1024 / 64, 16, 8), 128, 0, stream>>>(cqb, ckb, cvtb, attn, 1024, 256);
    gemm_bf16_kernel<EPI_RESADD><<<dim3(1024 / 256, 8192 / 128), 256, 0, stream>>>(
        attn, wcoT, b_co, xa, outP, nullptr, 8192, 1024, 1024);

    // 5) FFN (exact GELU), residual into d_out
    ln_mod_kernel<<<8192, 256, 0, stream>>>(outP, ada, hb, 1024, 4096, 5120);
    gemm_bf16_kernel<EPI_GELU><<<dim3(4096 / 256, 8192 / 128), 256, 0, stream>>>(
        hb, w1T, b1, nullptr, nullptr, g, 8192, 4096, 1024);
    gemm_bf16_kernel<EPI_RESADD><<<dim3(1024 / 256, 8192 / 128), 256, 0, stream>>>(
        g, w2T, b2, outP, outP, nullptr, 8192, 1024, 4096);
}
